// Network_76347338653985
// MI455X (gfx1250) — compile-verified, matrix operand order checked by reading, not run on hardware
//
#include <hip/hip_runtime.h>

#define N_NODES 16384
#define E_EDGES 262144
#define D_DIM   1024
#define DESM    1280
#define L_OUT   5000

#define BM 128
#define BN 64
#define BK 64
#define BKP 80   // padded LDS row stride in ushorts (160B: bank spread, 16B aligned chunks)

typedef __attribute__((ext_vector_type(16))) __bf16 v16bf;
typedef __attribute__((ext_vector_type(2)))  __bf16 v2bf;
typedef __attribute__((ext_vector_type(8)))  float  v8f;

union Frag {
    uint4 q[2];
    v16bf v;
};

__device__ __forceinline__ unsigned short f2bf(float x) {
#if __has_builtin(__builtin_amdgcn_cvt_pk_bf16_f32)
    v2bf p = __builtin_amdgcn_cvt_pk_bf16_f32(x, 0.f);
    return (unsigned short)(__builtin_bit_cast(unsigned int, p) & 0xffffu);
#else
    union { float f; unsigned int u; } c;
    c.f = x;
    unsigned int u = c.u + 0x7FFFu + ((c.u >> 16) & 1u);
    return (unsigned short)(u >> 16);
#endif
}

__device__ __forceinline__ uint2 pack4(float4 v) {
#if __has_builtin(__builtin_amdgcn_cvt_pk_bf16_f32)
    uint2 r;
    r.x = __builtin_bit_cast(unsigned int, __builtin_amdgcn_cvt_pk_bf16_f32(v.x, v.y));
    r.y = __builtin_bit_cast(unsigned int, __builtin_amdgcn_cvt_pk_bf16_f32(v.z, v.w));
    return r;
#else
    uint2 r;
    r.x = (unsigned int)f2bf(v.x) | ((unsigned int)f2bf(v.y) << 16);
    r.y = (unsigned int)f2bf(v.z) | ((unsigned int)f2bf(v.w) << 16);
    return r;
#endif
}

__device__ __forceinline__ float bf_lo(unsigned int w) { return __builtin_bit_cast(float, w << 16); }
__device__ __forceinline__ float bf_hi(unsigned int w) { return __builtin_bit_cast(float, w & 0xffff0000u); }

// async gather of one 16B chunk: global (bf16 row data) -> LDS (ASYNCcnt tracked)
__device__ __forceinline__ void async_b128(unsigned ldsOff, const void* gaddr) {
    asm volatile("global_load_async_to_lds_b128 %0, %1, off"
                 :: "v"(ldsOff), "v"(gaddr) : "memory");
}
template <int N>
__device__ __forceinline__ void wait_async() {
    asm volatile("s_wait_asynccnt %0" :: "n"(N) : "memory");
}

// ---------------------------------------------------------------------------
// f32 -> bf16 bulk convert (pre-pass for weights / embedding table)
// ---------------------------------------------------------------------------
__global__ __launch_bounds__(256) void cvt_bf16_kernel(
    const float4* __restrict__ s, uint2* __restrict__ d, int n4)
{
    int i = blockIdx.x * blockDim.x + threadIdx.x;
    const int stride = gridDim.x * blockDim.x;
    for (; i < n4; i += stride) d[i] = pack4(s[i]);
}

// ---------------------------------------------------------------------------
// EmbeddingBag(sum) + bias2 + relu -> x2 (f32, into the res buffer)
// ---------------------------------------------------------------------------
__global__ __launch_bounds__(256) void bag_kernel(
    const float* __restrict__ iw, const int* __restrict__ idx,
    const int* __restrict__ off, const float* __restrict__ bias2,
    float* __restrict__ x2)
{
    const int m = blockIdx.x;
    const int c = threadIdx.x * 4;
    const int b0 = off[m], b1 = off[m + 1];
    float a0 = 0.f, a1 = 0.f, a2 = 0.f, a3 = 0.f;
    for (int j = b0; j < b1; ++j) {
        const int r = idx[j];
        float4 v = *(const float4*)(iw + (size_t)r * D_DIM + c);
        a0 += v.x; a1 += v.y; a2 += v.z; a3 += v.w;
    }
    float4 b = *(const float4*)(bias2 + c);
    float4 o;
    o.x = fmaxf(a0 + b.x, 0.f);
    o.y = fmaxf(a1 + b.y, 0.f);
    o.z = fmaxf(a2 + b.z, 0.f);
    o.w = fmaxf(a3 + b.w, 0.f);
    *(float4*)(x2 + (size_t)m * D_DIM + c) = o;
}

__global__ __launch_bounds__(256) void zero_kernel(float4* __restrict__ p, int n4)
{
    int i = blockIdx.x * blockDim.x + threadIdx.x;
    const int stride = gridDim.x * blockDim.x;
    const float4 z = make_float4(0.f, 0.f, 0.f, 0.f);
    for (; i < n4; i += stride) p[i] = z;
}

// ---------------------------------------------------------------------------
// edge scatter: res += h[src]*self_w ; ppi += h[src]*ppi_w   (h is bf16)
// ---------------------------------------------------------------------------
__global__ __launch_bounds__(256) void edge_kernel(
    const unsigned short* __restrict__ hbf, const int* __restrict__ src,
    const int* __restrict__ dst, const float* __restrict__ sw,
    const float* __restrict__ pw, float* __restrict__ res,
    float* __restrict__ ppi)
{
    const int e = blockIdx.x;
    const int s = src[e], d = dst[e];
    const float a = sw[e], b = pw[e];
    const int c = threadIdx.x * 4;
    uint2 raw = *(const uint2*)(hbf + (size_t)s * D_DIM + c);
    float m0 = bf_lo(raw.x), m1 = bf_hi(raw.x);
    float m2 = bf_lo(raw.y), m3 = bf_hi(raw.y);
    float* rp = res + (size_t)d * D_DIM + c;
    float* pp = ppi + (size_t)d * D_DIM + c;
    __hip_atomic_fetch_add(rp + 0, m0 * a, __ATOMIC_RELAXED, __HIP_MEMORY_SCOPE_AGENT);
    __hip_atomic_fetch_add(rp + 1, m1 * a, __ATOMIC_RELAXED, __HIP_MEMORY_SCOPE_AGENT);
    __hip_atomic_fetch_add(rp + 2, m2 * a, __ATOMIC_RELAXED, __HIP_MEMORY_SCOPE_AGENT);
    __hip_atomic_fetch_add(rp + 3, m3 * a, __ATOMIC_RELAXED, __HIP_MEMORY_SCOPE_AGENT);
    __hip_atomic_fetch_add(pp + 0, m0 * b, __ATOMIC_RELAXED, __HIP_MEMORY_SCOPE_AGENT);
    __hip_atomic_fetch_add(pp + 1, m1 * b, __ATOMIC_RELAXED, __HIP_MEMORY_SCOPE_AGENT);
    __hip_atomic_fetch_add(pp + 2, m2 * b, __ATOMIC_RELAXED, __HIP_MEMORY_SCOPE_AGENT);
    __hip_atomic_fetch_add(pp + 3, m3 * b, __ATOMIC_RELAXED, __HIP_MEMORY_SCOPE_AGENT);
}

// ---------------------------------------------------------------------------
// WMMA bf16 GEMM: C = epilogue( A[M][K] @ W[Nout][K]^T + bias )
//  MODE 0: A row = pe_bf[inputs[m]] (async gather)    -> h_bf = sm0*relu(+b_esm+bias1)+sm1*x2
//  MODE 1: A row = cat(ppi,res) f32 (convert+ds_store)-> h_bf = relu(+b_upd)
//  MODE 2: A row = cat(h_bf, pe_bf[tgt[m]]) (async)   -> out f32 = +b_out (guard n)
//  B tile always async bf16 (row clamped for n>=Nout; garbage cols never stored)
// block 128x64, 8 waves of 32x32 (2x2 wmma), BK=64 (2 wmma-k per stage),
// double-buffered LDS + ASYNCcnt pipelining (wait <= next-stage issue count)
// ---------------------------------------------------------------------------
template <int MODE>
__global__ __launch_bounds__(256) void gemm_kernel(
    const unsigned short* __restrict__ A0bf, const unsigned short* __restrict__ A1bf,
    const float* __restrict__ F0, const float* __restrict__ F1,
    const int* __restrict__ aidx,
    const unsigned short* __restrict__ Wbf,
    const float* __restrict__ bias, const float* __restrict__ bias1,
    const float* __restrict__ x2, const float* __restrict__ wmix,
    float* __restrict__ Cf, unsigned short* __restrict__ Cbf,
    int K, int Ksplit, int Nout)
{
    constexpr int AISSUE = (MODE == 1) ? 2 : 6;   // async ops issued per stage per thread
    __shared__ __align__(16) unsigned short As[2 * BM * BKP];
    __shared__ __align__(16) unsigned short Bs[2 * BN * BKP];

    const int tid   = threadIdx.x;
    const int mBase = blockIdx.x * BM;
    const int nBase = blockIdx.y * BN;

    // per-thread fixed chunk assignment: 8 chunks/row of 16B (BK=64)
    const int ar  = tid >> 3;              // A rows: ar, ar+32, ar+64, ar+96
    const int ack = (tid & 7) * 8;         // bf16 elems within row
    int idxr[4];
    if (MODE != 1) {
        #pragma unroll
        for (int it = 0; it < 4; ++it) idxr[it] = aidx[mBase + ar + it * 32];
    }
    const int br = tid >> 3;               // B rows: br, br+32
    int gn0 = nBase + br;      if (gn0 > Nout - 1) gn0 = Nout - 1;
    int gn1 = nBase + br + 32; if (gn1 > Nout - 1) gn1 = Nout - 1;
    const unsigned short* bRow0 = Wbf + (size_t)gn0 * K + ack;
    const unsigned short* bRow1 = Wbf + (size_t)gn1 * K + ack;

    const unsigned aBase = (unsigned)(size_t)&As[0];
    const unsigned bBase = (unsigned)(size_t)&Bs[0];

    auto stage = [&](int k0, int sel) {
        const unsigned aB = aBase + (unsigned)sel * (BM * BKP * 2u);
        const unsigned bB = bBase + (unsigned)sel * (BN * BKP * 2u);
        if (MODE == 0) {
            #pragma unroll
            for (int it = 0; it < 4; ++it) {
                const int row = ar + it * 32;
                const unsigned short* g = A0bf + (size_t)idxr[it] * DESM + k0 + ack;
                async_b128(aB + (unsigned)(row * BKP + ack) * 2u, g);
            }
        } else if (MODE == 2) {
            const int k = k0 + ack;
            #pragma unroll
            for (int it = 0; it < 4; ++it) {
                const int row = ar + it * 32;
                const unsigned short* g = (k < Ksplit)
                    ? (A0bf + (size_t)(mBase + row) * Ksplit + k)
                    : (A1bf + (size_t)idxr[it] * DESM + (k - Ksplit));
                async_b128(aB + (unsigned)(row * BKP + ack) * 2u, g);
            }
        } else {
            // f32 res/ppi -> cvt_pk bf16 -> LDS (2048 float4, 8/thread)
            unsigned short* AsBuf = &As[sel * BM * BKP];
            #pragma unroll
            for (int it = 0; it < 8; ++it) {
                const int f   = tid + it * 256;
                const int row = f >> 4;
                const int c4  = (f & 15) * 4;
                const int k   = k0 + c4;
                const float* src = (k < Ksplit)
                    ? (F0 + (size_t)(mBase + row) * Ksplit + k)
                    : (F1 + (size_t)(mBase + row) * Ksplit + (k - Ksplit));
                float4 v = *(const float4*)src;
                if (k0 + BK < K) __builtin_prefetch(src + BK, 0, 1);
                *(uint2*)&AsBuf[row * BKP + c4] = pack4(v);
            }
        }
        async_b128(bB + (unsigned)(br * BKP + ack) * 2u, bRow0 + k0);
        async_b128(bB + (unsigned)((br + 32) * BKP + ack) * 2u, bRow1 + k0);
    };

    v8f acc[2][2] = {};
    const int lane = tid & 31;
    const int wave = tid >> 5;
    const int wm   = (wave >> 1) * 32;
    const int wn   = (wave & 1)  * 32;
    const int half = lane >> 4;
    const int l    = lane & 15;

    const int nSteps = K / BK;
    stage(0, 0);
    for (int s = 0; s < nSteps; ++s) {
        const int sel = s & 1;
        if (s + 1 < nSteps) {
            stage((s + 1) * BK, sel ^ 1);     // lookahead into other buffer
            wait_async<AISSUE>();             // stage s resident (in-order completion)
        } else {
            wait_async<0>();
        }
        __syncthreads();                      // stage s visible to all waves

        const unsigned short* AsBuf = &As[sel * BM * BKP];
        const unsigned short* BsBuf = &Bs[sel * BN * BKP];
        #pragma unroll
        for (int kk = 0; kk < BK; kk += 32) {
            const int kLo = kk + half * 8;
            Frag a[2], b[2];
            #pragma unroll
            for (int i = 0; i < 2; ++i) {
                const unsigned short* p = &AsBuf[(wm + i * 16 + l) * BKP + kLo];
                a[i].q[0] = *(const uint4*)p;
                a[i].q[1] = *(const uint4*)(p + 16);
            }
            #pragma unroll
            for (int j = 0; j < 2; ++j) {
                const unsigned short* p = &BsBuf[(wn + j * 16 + l) * BKP + kLo];
                b[j].q[0] = *(const uint4*)p;
                b[j].q[1] = *(const uint4*)(p + 16);
            }
            #pragma unroll
            for (int i = 0; i < 2; ++i)
                #pragma unroll
                for (int j = 0; j < 2; ++j)
                    acc[i][j] = __builtin_amdgcn_wmma_f32_16x16x32_bf16(
                        false, a[i].v, false, b[j].v, (short)0, acc[i][j], false, false);
        }
        __syncthreads();   // readers done before next iteration overwrites this buffer
    }

    // ---- epilogue ----
    float sm0 = 0.f, sm1 = 0.f;
    if (MODE == 0) {
        const float w0 = wmix[0], w1 = wmix[1];
        const float mx = fmaxf(w0, w1);
        const float e0 = __expf(w0 - mx), e1 = __expf(w1 - mx);
        const float inv = 1.f / (e0 + e1);
        sm0 = e0 * inv; sm1 = e1 * inv;
    }

    #pragma unroll
    for (int i = 0; i < 2; ++i) {
        #pragma unroll
        for (int j = 0; j < 2; ++j) {
            const int gn = nBase + wn + j * 16 + l;
            if (MODE == 2 && gn >= Nout) continue;
            const float bv  = bias[gn];
            const float b1v = (MODE == 0) ? bias1[gn] : 0.f;
            #pragma unroll
            for (int r = 0; r < 8; ++r) {
                const int gm = mBase + wm + i * 16 + half * 8 + r;
                float v = acc[i][j][r] + bv;
                if (MODE == 0) {
                    v = fmaxf(v + b1v, 0.f);
                    v = sm0 * v + sm1 * x2[(size_t)gm * D_DIM + gn];
                    Cbf[(size_t)gm * D_DIM + gn] = f2bf(v);
                } else if (MODE == 1) {
                    v = fmaxf(v, 0.f);
                    Cbf[(size_t)gm * D_DIM + gn] = f2bf(v);
                } else {
                    Cf[(size_t)gm * Nout + gn] = v;
                }
            }
        }
    }
}

// ---------------------------------------------------------------------------
extern "C" void kernel_launch(void* const* d_in, const int* in_sizes, int n_in,
                              void* d_out, int out_size, void* d_ws, size_t ws_size,
                              hipStream_t stream) {
    const float* pe     = (const float*)d_in[0];
    const float* iw     = (const float*)d_in[1];
    const float* W_esm  = (const float*)d_in[2];
    const float* b_esm  = (const float*)d_in[3];
    const float* bias1  = (const float*)d_in[4];
    const float* bias2  = (const float*)d_in[5];
    const float* wmix   = (const float*)d_in[6];
    const float* W_upd  = (const float*)d_in[7];
    const float* b_upd  = (const float*)d_in[8];
    const float* W_out  = (const float*)d_in[9];
    const float* b_out  = (const float*)d_in[10];
    const float* self_w = (const float*)d_in[11];
    const float* ppi_w  = (const float*)d_in[12];
    const int*   inputs = (const int*)d_in[13];
    const int*   ipidx  = (const int*)d_in[14];
    const int*   ipoff  = (const int*)d_in[15];
    const int*   src    = (const int*)d_in[16];
    const int*   dst    = (const int*)d_in[17];
    const int*   tgt    = (const int*)d_in[18];

    char* ws = (char*)d_ws;
    float*          res     = (float*)ws;                                  // also x2
    float*          ppi     = (float*)(ws + 67108864);
    unsigned short* h_bf    = (unsigned short*)(ws + 134217728);
    unsigned short* pe_bf   = (unsigned short*)(ws + 167772160);
    unsigned short* wesm_bf = (unsigned short*)(ws + 218972160);
    unsigned short* wupd_bf = (unsigned short*)(ws + 221593600);
    unsigned short* wout_bf = (unsigned short*)(ws + 229982208);
    float*          out     = (float*)d_out;

    // 0) pre-convert reused f32 operands to bf16
    cvt_bf16_kernel<<<2048, 256, 0, stream>>>((const float4*)pe,    (uint2*)pe_bf,   20000 * DESM / 4);
    cvt_bf16_kernel<<< 512, 256, 0, stream>>>((const float4*)W_esm, (uint2*)wesm_bf, D_DIM * DESM / 4);
    cvt_bf16_kernel<<<1024, 256, 0, stream>>>((const float4*)W_upd, (uint2*)wupd_bf, 2 * D_DIM * 2 * D_DIM / 4);
    cvt_bf16_kernel<<<2048, 256, 0, stream>>>((const float4*)W_out, (uint2*)wout_bf, L_OUT * (D_DIM + DESM) / 4);

    // 1) x2 = relu(EmbeddingBag + bias2) -> res buffer (free at this point)
    bag_kernel<<<N_NODES, 256, 0, stream>>>(iw, ipidx, ipoff, bias2, res);

    // 2) h_bf = sm0*relu(pe[inputs] @ W_esm^T + b_esm + bias1) + sm1*x2
    dim3 g1(N_NODES / BM, D_DIM / BN);
    gemm_kernel<0><<<g1, 256, 0, stream>>>(pe_bf, nullptr, nullptr, nullptr, inputs,
                                           wesm_bf, b_esm, bias1, res, wmix,
                                           nullptr, h_bf, DESM, 0, D_DIM);

    // 3) two GCN layers
    const int n4 = 2 * N_NODES * D_DIM / 4;
    for (int i = 0; i < 2; ++i) {
        zero_kernel<<<4096, 256, 0, stream>>>((float4*)res, n4);          // res+ppi contiguous
        edge_kernel<<<E_EDGES, 256, 0, stream>>>(h_bf, src, dst, self_w, ppi_w, res, ppi);
        gemm_kernel<1><<<g1, 256, 0, stream>>>(nullptr, nullptr, ppi, res, nullptr,
                                               wupd_bf + (size_t)i * D_DIM * 2 * D_DIM,
                                               b_upd + (size_t)i * D_DIM,
                                               nullptr, nullptr, nullptr,
                                               nullptr, h_bf, 2 * D_DIM, D_DIM, D_DIM);
    }

    // 4) out = cat(h, pe[target]) @ W_out^T + b_out
    dim3 g2(N_NODES / BM, (L_OUT + BN - 1) / BN);
    gemm_kernel<2><<<g2, 256, 0, stream>>>(h_bf, pe_bf, nullptr, nullptr, tgt,
                                           wout_bf, b_out, nullptr, nullptr, nullptr,
                                           out, nullptr, D_DIM + DESM, D_DIM, L_OUT);
}
